// TransformerBlock_46643344835182
// MI455X (gfx1250) — compile-verified, matrix operand order checked by reading, not run on hardware
//
#include <hip/hip_runtime.h>
#include <hip/hip_bf16.h>
#include <math.h>

// ---------------------------------------------------------------------------
// GPT-2 transformer block for MI455X (gfx1250, wave32, WMMA).
// bf16 matrix inputs + f32 WMMA accumulation via v_wmma_f32_16x16x32_bf16.
// GEMM A-tiles staged with GLOBAL_LOAD_ASYNC_TO_LDS_B128 (ASYNCcnt path).
// ---------------------------------------------------------------------------

typedef __bf16 bf16_t;
typedef __attribute__((ext_vector_type(16))) __bf16 v16bf;
typedef __attribute__((ext_vector_type(8)))  float  v8f;

#define D_MODEL 768
#define N_HEADS 12
#define D_HEADD 64
#define D_MLP   3072
#define SEQ     1024
#define BATCH   8
#define ROWS    (BATCH * SEQ)   // 8192
#define NQKV    (3 * D_MODEL)   // 2304

// ---- fragment gather (ISA 7.12.2) ------------------------------------------
// 16-bit A-matrix 16x32: lane L (r=L&15, g=L>>4) holds row M=r;
// element e of the v16 fragment is K = (e>=8?16:0) + 8*g + (e&7).
// B fragments use the same mapping applied to the n-major (transposed) tile.
static __device__ inline v16bf ldfrag(const bf16_t* base, int ld, int lane) {
  const int r = lane & 15, g = lane >> 4;
  const bf16_t* p = base + (size_t)r * ld + g * 8;
  v16bf f;
#pragma unroll
  for (int e = 0; e < 16; ++e)
    f[e] = p[((e & 8) << 1) + (e & 7)];
  return f;
}

static __device__ inline v8f wmma_bf16(v16bf a, v16bf b, v8f c) {
  // (neg_a, A, neg_b, B, c_mod, C, reuse_a, reuse_b)
  return __builtin_amdgcn_wmma_f32_16x16x32_bf16(false, a, false, b, (short)0,
                                                 c, false, false);
}

// async global->LDS copy of 16 bytes per lane (tracked by ASYNCcnt)
static __device__ inline void async_copy_b128(const bf16_t* gsrc, void* lds_dst) {
  const unsigned lds_off = (unsigned)(uintptr_t)lds_dst;  // addr[31:0] = LDS offset
  const unsigned long long gaddr = (unsigned long long)(uintptr_t)gsrc;
  asm volatile("global_load_async_to_lds_b128 %0, %1, off"
               :: "v"(lds_off), "v"(gaddr) : "memory");
}
static __device__ inline void wait_asynccnt0() {
  asm volatile("s_wait_asynccnt 0x0" ::: "memory");
}

// reductions across a 16-lane half-wave (C-layout rows live in one half)
static __device__ inline float hmax16(float v) {
#pragma unroll
  for (int m = 1; m <= 8; m <<= 1) v = fmaxf(v, __shfl_xor(v, m, 32));
  return v;
}
static __device__ inline float hsum16(float v) {
#pragma unroll
  for (int m = 1; m <= 8; m <<= 1) v += __shfl_xor(v, m, 32);
  return v;
}

static __device__ inline float gelu_new_f(float x) {
  const float c = 0.7978845608028654f;  // sqrt(2/pi)
  return 0.5f * x * (1.0f + tanhf(c * (x + 0.044715f * x * x * x)));
}

// ---------------------------------------------------------------------------
// Generic WMMA GEMM: C[M,N] = A[M,K](bf16,row-major) * B[K,N](bf16,row-major)
// + bias[col], optional gelu, optional fp32 residual add, f32 or bf16 out.
// Block = 128 threads (4 waves), 64x64 output tile, K stepped by 32.
// ---------------------------------------------------------------------------
template <int GELU, int RESID, int OUTBF>
__global__ __launch_bounds__(128) void gemm_bf16_k(
    const bf16_t* __restrict__ A, const bf16_t* __restrict__ B,
    const float* __restrict__ bias, const float* __restrict__ resid,
    float* __restrict__ outF, bf16_t* __restrict__ outB, int K, int N) {
  __shared__ __align__(16) bf16_t As[64][48];  // row-major, 96B row stride
  __shared__ __align__(16) bf16_t Bt[64][48];  // n-major (transposed) tile
  const int tid = threadIdx.x;
  const int lane = tid & 31, wave = tid >> 5;
  const int n0 = blockIdx.x * 64, m0 = blockIdx.y * 64;
  v8f acc[4] = {{}, {}, {}, {}};

  for (int k0 = 0; k0 < K; k0 += 32) {
    if (k0 + 32 < K) {  // prefetch next tiles -> global_prefetch_b8
      __builtin_prefetch(A + (size_t)(m0 + (tid >> 1)) * K + k0 + 32, 0, 1);
      __builtin_prefetch(B + (size_t)(k0 + 32 + (tid >> 2)) * N + n0, 0, 1);
    }
    // stage A 64x32 via async global->LDS b128 copies (no VGPR bounce)
    for (int c = tid; c < 256; c += 128) {
      const int row = c >> 2, cg = c & 3;
      async_copy_b128(A + (size_t)(m0 + row) * K + k0 + cg * 8,
                      &As[row][cg * 8]);
    }
    // stage B 32x64 transposed -> Bt[n][k]
    for (int i = tid; i < 2048; i += 128) {
      const int n = i & 63, k = i >> 6;
      Bt[n][k] = B[(size_t)(k0 + k) * N + n0 + n];
    }
    wait_asynccnt0();  // async A-tile landed in LDS
    __syncthreads();
    const v16bf aF = ldfrag(&As[wave * 16][0], 48, lane);
#pragma unroll
    for (int nb = 0; nb < 4; ++nb) {
      const v16bf bF = ldfrag(&Bt[nb * 16][0], 48, lane);
      acc[nb] = wmma_bf16(aF, bF, acc[nb]);
    }
    __syncthreads();
  }

  // C layout: VGPR j holds row 8*g + j, column = lane&15 within the 16-block
  const int r = lane & 15, g = lane >> 4;
  const int row0 = m0 + wave * 16 + g * 8;
#pragma unroll
  for (int nb = 0; nb < 4; ++nb) {
    const int col = n0 + nb * 16 + r;
    const float bv = bias[col];
#pragma unroll
    for (int j = 0; j < 8; ++j) {
      const int row = row0 + j;
      float v = acc[nb][j] + bv;
      if (GELU) v = gelu_new_f(v);
      if (RESID) v += resid[(size_t)row * N + col];
      if (OUTBF) outB[(size_t)row * N + col] = (bf16_t)v;
      else       outF[(size_t)row * N + col] = v;
    }
  }
}

// ---------------------------------------------------------------------------
// Flash-style causal attention: one wave per (batch, head, 16-query tile).
// Scores and PV both through v_wmma; online softmax in fp32.
// qkv rows are [8192, 2304] = [Q(12*64) | K(12*64) | V(12*64)].
// ---------------------------------------------------------------------------
__global__ __launch_bounds__(32) void attn_flash_k(
    const bf16_t* __restrict__ qkv, bf16_t* __restrict__ zbuf) {
  __shared__ __align__(16) bf16_t Vt[64][34];  // V^T tile: Vt[e][kk]
  __shared__ __align__(16) bf16_t Ps[16][34];  // probabilities 16x32
  const int lane = threadIdx.x;
  const int r = lane & 15, g = lane >> 4;
  const int qbase = blockIdx.x * 16;
  const int h = blockIdx.y, b = blockIdx.z;
  const bf16_t* Qp = qkv + (size_t)b * SEQ * NQKV + h * D_HEADD;
  const bf16_t* Kp = Qp + D_MODEL;
  const bf16_t* Vp = Qp + 2 * D_MODEL;

  // Q tile fragments (16 rows x 64 e), two K-steps of 32
  const v16bf aQ0 = ldfrag(Qp + (size_t)qbase * NQKV, NQKV, lane);
  const v16bf aQ1 = ldfrag(Qp + (size_t)qbase * NQKV + 32, NQKV, lane);

  float m[8], l[8];
  v8f acc[4] = {{}, {}, {}, {}};
#pragma unroll
  for (int j = 0; j < 8; ++j) { m[j] = -1e30f; l[j] = 0.0f; }

  const int kbmax = (qbase + 15) >> 5;  // inclusive 32-key chunks
  for (int kb = 0; kb <= kbmax; ++kb) {
    const int ks = kb * 32;
    // stage V^T (lane owns key row ks+lane)
    const bf16_t* vrow = Vp + (size_t)(ks + lane) * NQKV;
#pragma unroll
    for (int e = 0; e < 64; ++e) Vt[e][lane] = vrow[e];

    // scores: S = Q K^T. K memory [s][e] is already n-major for the B frag.
    v8f S0 = {}, S1 = {};
    S0 = wmma_bf16(aQ0, ldfrag(Kp + (size_t)ks * NQKV, NQKV, lane), S0);
    S0 = wmma_bf16(aQ1, ldfrag(Kp + (size_t)ks * NQKV + 32, NQKV, lane), S0);
    S1 = wmma_bf16(aQ0, ldfrag(Kp + (size_t)(ks + 16) * NQKV, NQKV, lane), S1);
    S1 = wmma_bf16(aQ1, ldfrag(Kp + (size_t)(ks + 16) * NQKV + 32, NQKV, lane), S1);

    float p0[8], p1[8], al[8];
#pragma unroll
    for (int j = 0; j < 8; ++j) {
      const int qg = qbase + 8 * g + j;
      float s0 = S0[j] * 0.125f; if (ks + r > qg)      s0 = -1e30f;  // causal
      float s1 = S1[j] * 0.125f; if (ks + 16 + r > qg) s1 = -1e30f;
      const float mn = fmaxf(m[j], hmax16(fmaxf(s0, s1)));
      al[j] = __expf(m[j] - mn);
      p0[j] = __expf(s0 - mn);
      p1[j] = __expf(s1 - mn);
      l[j] = l[j] * al[j] + hsum16(p0[j] + p1[j]);
      m[j] = mn;
    }
#pragma unroll
    for (int nb = 0; nb < 4; ++nb)
#pragma unroll
      for (int j = 0; j < 8; ++j) acc[nb][j] *= al[j];

    // restage P through LDS into A-fragment layout
#pragma unroll
    for (int j = 0; j < 8; ++j) {
      Ps[8 * g + j][r]      = (bf16_t)p0[j];
      Ps[8 * g + j][16 + r] = (bf16_t)p1[j];
    }
    asm volatile("s_wait_dscnt 0" ::: "memory");  // same-wave LDS RAW
    const v16bf aP = ldfrag(&Ps[0][0], 34, lane);
#pragma unroll
    for (int nb = 0; nb < 4; ++nb) {
      const v16bf bV = ldfrag(&Vt[nb * 16][0], 34, lane);
      acc[nb] = wmma_bf16(aP, bV, acc[nb]);
    }
  }

  // z = acc / l  -> zbuf[8192, 768] at column h*64
  bf16_t* zr = zbuf + (size_t)(b * SEQ + qbase + 8 * g) * D_MODEL + h * D_HEADD;
#pragma unroll
  for (int nb = 0; nb < 4; ++nb)
#pragma unroll
    for (int j = 0; j < 8; ++j)
      zr[(size_t)j * D_MODEL + nb * 16 + r] = (bf16_t)(acc[nb][j] / l[j]);
}

// ---------------------------------------------------------------------------
// LayerNorm (fp32 in) -> bf16 out. One 256-thread block per row (768 elems).
// ---------------------------------------------------------------------------
__global__ __launch_bounds__(256) void ln_bf16_k(
    const float* __restrict__ x, const float* __restrict__ w,
    const float* __restrict__ bb, bf16_t* __restrict__ out) {
  __shared__ float red[256];
  const int row = blockIdx.x, tid = threadIdx.x;
  const float* xr = x + (size_t)row * D_MODEL;
  const float v0 = xr[tid], v1 = xr[tid + 256], v2 = xr[tid + 512];
  red[tid] = v0 + v1 + v2;
  __syncthreads();
  for (int off = 128; off > 0; off >>= 1) {
    if (tid < off) red[tid] += red[tid + off];
    __syncthreads();
  }
  const float mu = red[0] * (1.0f / 768.0f);
  __syncthreads();
  const float d0 = v0 - mu, d1 = v1 - mu, d2 = v2 - mu;
  red[tid] = d0 * d0 + d1 * d1 + d2 * d2;
  __syncthreads();
  for (int off = 128; off > 0; off >>= 1) {
    if (tid < off) red[tid] += red[tid + off];
    __syncthreads();
  }
  const float rs = rsqrtf(red[0] * (1.0f / 768.0f) + 1e-5f);
  bf16_t* orow = out + (size_t)row * D_MODEL;
  orow[tid]       = (bf16_t)(d0 * rs * w[tid]       + bb[tid]);
  orow[tid + 256] = (bf16_t)(d1 * rs * w[tid + 256] + bb[tid + 256]);
  orow[tid + 512] = (bf16_t)(d2 * rs * w[tid + 512] + bb[tid + 512]);
}

// ---- weight prep -----------------------------------------------------------
__global__ void cvt_f32_bf16_k(const float* __restrict__ in,
                               bf16_t* __restrict__ out, int n) {
  const int i = blockIdx.x * 256 + threadIdx.x;
  if (i < n) out[i] = (bf16_t)in[i];
}
// W_{Q,K,V}[h,d,e] -> Wqkv[d, sec*768 + h*64 + e]  (bf16, [768,2304])
__global__ void pack_qkv_w_k(const float* __restrict__ Wq,
                             const float* __restrict__ Wk,
                             const float* __restrict__ Wv,
                             bf16_t* __restrict__ out) {
  const int i = blockIdx.x * 256 + threadIdx.x;
  if (i >= N_HEADS * D_MODEL * D_HEADD) return;
  const int h = i / (D_MODEL * D_HEADD);
  const int rem = i - h * (D_MODEL * D_HEADD);
  const int d = rem / D_HEADD, e = rem - d * D_HEADD;
  const size_t o = (size_t)d * NQKV + h * D_HEADD + e;
  out[o]               = (bf16_t)Wq[i];
  out[o + D_MODEL]     = (bf16_t)Wk[i];
  out[o + 2 * D_MODEL] = (bf16_t)Wv[i];
}
__global__ void pack_qkv_b_k(const float* __restrict__ q,
                             const float* __restrict__ k,
                             const float* __restrict__ v,
                             float* __restrict__ out) {
  const int i = blockIdx.x * 256 + threadIdx.x;
  if (i < D_MODEL) {
    out[i] = q[i];
    out[i + D_MODEL] = k[i];
    out[i + 2 * D_MODEL] = v[i];
  }
}

// ---------------------------------------------------------------------------
extern "C" void kernel_launch(void* const* d_in, const int* in_sizes, int n_in,
                              void* d_out, int out_size, void* d_ws,
                              size_t ws_size, hipStream_t stream) {
  (void)in_sizes; (void)n_in; (void)out_size; (void)ws_size;
  const float* resid_pre = (const float*)d_in[0];
  const float* ln1_w = (const float*)d_in[1];
  const float* ln1_b = (const float*)d_in[2];
  const float* W_Q = (const float*)d_in[3];
  const float* b_Q = (const float*)d_in[4];
  const float* W_K = (const float*)d_in[5];
  const float* b_K = (const float*)d_in[6];
  const float* W_V = (const float*)d_in[7];
  const float* b_V = (const float*)d_in[8];
  const float* W_O = (const float*)d_in[9];   // [12,64,768] == row-major [768,768]
  const float* b_O = (const float*)d_in[10];
  const float* ln2_w = (const float*)d_in[11];
  const float* ln2_b = (const float*)d_in[12];
  const float* W_in = (const float*)d_in[13];  // [768,3072]
  const float* b_in = (const float*)d_in[14];
  const float* W_out = (const float*)d_in[15]; // [3072,768]
  const float* b_out = (const float*)d_in[16];

  char* ws = (char*)d_ws;
  size_t off = 0;
  auto take = [&](size_t bytes) {
    char* p = ws + off;
    off += (bytes + 255) & ~(size_t)255;
    return p;
  };
  bf16_t* xn1  = (bf16_t*)take((size_t)ROWS * D_MODEL * 2);   // also xn2
  bf16_t* wqkv = (bf16_t*)take((size_t)D_MODEL * NQKV * 2);
  float*  bqkv = (float*) take((size_t)NQKV * 4);
  bf16_t* qkv  = (bf16_t*)take((size_t)ROWS * NQKV * 2);      // hbuf aliases here
  bf16_t* zbuf = (bf16_t*)take((size_t)ROWS * D_MODEL * 2);
  bf16_t* wo   = (bf16_t*)take((size_t)D_MODEL * D_MODEL * 2);
  float*  res  = (float*) take((size_t)ROWS * D_MODEL * 4);
  bf16_t* win  = (bf16_t*)take((size_t)D_MODEL * D_MLP * 2);
  bf16_t* wout = (bf16_t*)take((size_t)D_MLP * D_MODEL * 2);
  bf16_t* xn2  = xn1;          // dead after QKV GEMM
  bf16_t* hbuf = qkv;          // qkv+zbuf region (50.3MB) dead after O-proj

  // weight / bias prep (bf16 packing)
  pack_qkv_w_k<<<(N_HEADS * D_MODEL * D_HEADD + 255) / 256, 256, 0, stream>>>(
      W_Q, W_K, W_V, wqkv);
  pack_qkv_b_k<<<3, 256, 0, stream>>>(b_Q, b_K, b_V, bqkv);
  cvt_f32_bf16_k<<<(D_MODEL * D_MODEL + 255) / 256, 256, 0, stream>>>(
      W_O, wo, D_MODEL * D_MODEL);
  cvt_f32_bf16_k<<<(D_MODEL * D_MLP + 255) / 256, 256, 0, stream>>>(
      W_in, win, D_MODEL * D_MLP);
  cvt_f32_bf16_k<<<(D_MLP * D_MODEL + 255) / 256, 256, 0, stream>>>(
      W_out, wout, D_MLP * D_MODEL);

  // LN1 -> xn1 (bf16)
  ln_bf16_k<<<ROWS, 256, 0, stream>>>(resid_pre, ln1_w, ln1_b, xn1);

  // fused QKV projection: xn1[8192,768] x wqkv[768,2304] + bias -> qkv (bf16)
  gemm_bf16_k<0, 0, 1><<<dim3(NQKV / 64, ROWS / 64), 128, 0, stream>>>(
      xn1, wqkv, bqkv, nullptr, nullptr, qkv, D_MODEL, NQKV);

  // flash causal attention -> zbuf[8192,768] (bf16)
  attn_flash_k<<<dim3(SEQ / 16, N_HEADS, BATCH), 32, 0, stream>>>(qkv, zbuf);

  // O-projection + residual: zbuf x W_O + b_O + resid_pre -> res (f32)
  gemm_bf16_k<0, 1, 0><<<dim3(D_MODEL / 64, ROWS / 64), 128, 0, stream>>>(
      zbuf, wo, b_O, resid_pre, res, nullptr, D_MODEL, D_MODEL);

  // LN2 -> xn2 (bf16)
  ln_bf16_k<<<ROWS, 256, 0, stream>>>(res, ln2_w, ln2_b, xn2);

  // MLP in + gelu: xn2 x W_in + b_in -> hbuf (bf16)
  gemm_bf16_k<1, 0, 1><<<dim3(D_MLP / 64, ROWS / 64), 128, 0, stream>>>(
      xn2, win, b_in, nullptr, nullptr, hbuf, D_MODEL, D_MLP);

  // MLP out + residual: hbuf x W_out + b_out + res -> d_out (f32)
  gemm_bf16_k<0, 1, 0><<<dim3(D_MODEL / 64, ROWS / 64), 128, 0, stream>>>(
      hbuf, wout, b_out, res, (float*)d_out, nullptr, D_MLP, D_MODEL);
}
